// ChemotaxisPINN_64854006169661
// MI455X (gfx1250) — matrix operand order repaired and checked by reading, not compiled, not run on hardware
//
#include <hip/hip_runtime.h>

#define HDIM 128
#define NWAVES 4
#define TPB (NWAVES * 32)

typedef _Float16 v16h __attribute__((ext_vector_type(16)));
typedef _Float16 v8h  __attribute__((ext_vector_type(8)));
typedef float    v8f  __attribute__((ext_vector_type(8)));

union V16 { v16h v; v8h h[2]; };

static constexpr int CN_COL = 131072;
static constexpr int CN_IC  = 2000;
static constexpr int CN_BC  = 2000;

// LDS layout (dynamic) -- Stage FIRST so every staging address fits the
// 16-bit DS immediate-offset field (stage spans exactly [0, 65536)):
//  [0)        Stage: [NW][4 streams][16 pts][128 feat] f16      (65536 B)
//  [65536)    Wt: 3 layers of W^T as f16, [layer][n][k]         (98304 B)
//  [163840)   floats: Bs[3*128], W0r[2*128], B0s[128], W4s[128], PX, PT
static constexpr int SMEM_STAGE_BYTES = NWAVES * 4 * 16 * HDIM * 2;           // 65536
static constexpr int SMEM_WT_BYTES    = 3 * HDIM * HDIM * 2;                  // 98304
static constexpr int SMEM_FLT         = 3*HDIM + 2*HDIM + HDIM + HDIM + NWAVES*16 + NWAVES*16; // 1024
static constexpr size_t SMEM_BYTES    = (size_t)SMEM_STAGE_BYTES + (size_t)SMEM_WT_BYTES + (size_t)SMEM_FLT*4;

__device__ __forceinline__ v8f wmma16(v16h a, v16h b, v8f c) {
  // D = A(16x32 f16) * B(32x16 f16) + C(16x16 f32)
  return __builtin_amdgcn_wmma_f32_16x16x32_f16(false, a, false, b, (short)0, c, false, false);
}

// Fast sigmoid: one v_exp_f32 + one v_rcp_f32 (no IEEE div chain).
__device__ __forceinline__ float fsigmoid(float z) {
  float e = __builtin_amdgcn_exp2f(z * -1.44269504088896340736f); // exp(-z)
  return __builtin_amdgcn_rcpf(1.f + e);
}

// MODE 0: PDE residual   r = u_t - D u_xx + chi*(u_x*S'(x) + u*S''(x))
// MODE 1: IC residual    r = u - ic(x)
// MODE 2: BC residual    r = u_x
template<int MODE>
__global__ void __launch_bounds__(TPB)
pinn_kernel(const float* __restrict__ X, const float* __restrict__ T, int npts,
            const float* __restrict__ W0, const float* __restrict__ b0,
            const float* __restrict__ W1, const float* __restrict__ b1,
            const float* __restrict__ W2, const float* __restrict__ b2,
            const float* __restrict__ W3, const float* __restrict__ b3,
            const float* __restrict__ W4, const float* __restrict__ b4,
            const float* __restrict__ Dp, const float* __restrict__ chip,
            float* __restrict__ accum) {
  extern __shared__ __align__(16) char smem[];
  _Float16* Stage = (_Float16*)smem;                               // [NW][4][16][128]
  _Float16* Wt  = (_Float16*)(smem + SMEM_STAGE_BYTES);            // [3][128][128]
  float*    Bs  = (float*)(smem + SMEM_STAGE_BYTES + SMEM_WT_BYTES); // [3][128]
  float*    W0r = Bs  + 3*HDIM;                                    // [2][128]
  float*    B0s = W0r + 2*HDIM;                                    // [128]
  float*    W4s = B0s + HDIM;                                      // [128]
  float*    PX  = W4s + HDIM;                                      // [NWAVES*16]
  float*    PT  = PX  + NWAVES*16;                                 // [NWAVES*16]

  const int tid  = threadIdx.x;
  const int wave = tid >> 5;
  const int lane = tid & 31;
  const int lhalf= lane >> 4;   // 0 or 1
  const int l16  = lane & 15;

  // ---- preload weights to LDS (W1..W3 transposed, converted to f16) ----
  {
    const float* Wl[3] = { W1, W2, W3 };
    const float* bl[3] = { b1, b2, b3 };
    for (int layer = 0; layer < 3; ++layer) {
      const float* W = Wl[layer];
      for (int i = tid; i < HDIM*HDIM; i += TPB) {
        int n = i / HDIM, k = i % HDIM;                 // W is [k][n] row-major
        Wt[layer*HDIM*HDIM + n*HDIM + k] = (_Float16)W[k*HDIM + n];
      }
      for (int i = tid; i < HDIM; i += TPB) Bs[layer*HDIM + i] = bl[layer][i];
    }
    for (int i = tid; i < 2*HDIM; i += TPB) W0r[i] = W0[i];
    for (int i = tid; i < HDIM;   i += TPB) { B0s[i] = b0[i]; W4s[i] = W4[i]; }
  }
  __syncthreads();   // only cross-wave barrier needed: everything below is per-wave

  const float Dv = Dp[0], chiv = chip[0], b4v = b4[0];

  _Float16* Sh  = Stage + (wave*4 + 0) * 16*HDIM;
  _Float16* Sx  = Stage + (wave*4 + 1) * 16*HDIM;
  _Float16* St  = Stage + (wave*4 + 2) * 16*HDIM;
  _Float16* Sxx = Stage + (wave*4 + 3) * 16*HDIM;

  const int tileSpan = NWAVES * 16;
  const int npad = ((npts + tileSpan - 1) / tileSpan) * tileSpan;

  for (int base = blockIdx.x * tileSpan; base < npad; base += gridDim.x * tileSpan) {
    const int m0 = base + wave * 16;

    // ---- load this wave's 16 points (same-wave DS ordering makes this safe) ----
    if (lane < 16) {
      int idx = m0 + lane;
      float xv = 0.f, tv = 0.f;
      if (idx < npts) { xv = X[idx]; tv = T[idx]; }
      PX[wave*16 + lane] = xv;
      PT[wave*16 + lane] = tv;
    }

    // ---- layer 0 (2 -> 128), VALU, write staging in [m][k] layout ----
    for (int e = lane; e < 16*HDIM; e += 32) {
      int m = e >> 7, n = e & (HDIM-1);
      float xv = PX[wave*16 + m], tv = PT[wave*16 + m];
      float w0 = W0r[n], w1 = W0r[HDIM + n];
      float z  = xv*w0 + tv*w1 + B0s[n];
      float sg = fsigmoid(z);
      float d1 = sg * (1.f + z*(1.f - sg));
      float d2 = sg * (1.f - sg) * (2.f + z*(1.f - 2.f*sg));
      Sh [e] = (_Float16)(z * sg);
      Sx [e] = (_Float16)(d1 * w0);
      St [e] = (_Float16)(d1 * w1);
      Sxx[e] = (_Float16)(d2 * w0 * w0);
    }

    // ---- layers 1..3: WMMA on 4 streams ----
    for (int layer = 0; layer < 3; ++layer) {
      // Load A fragments: 4 streams x 4 K-chunks; lane holds row M=l16,
      // halves 0..7 -> K = 32c + lhalf*8 .. +7 ; halves 8..15 -> +16.
      V16 A[4][4];
      #pragma unroll
      for (int s = 0; s < 4; ++s) {
        const _Float16* S = Stage + (wave*4 + s)*16*HDIM + l16*HDIM;
        #pragma unroll
        for (int c = 0; c < 4; ++c) {
          int kb = 32*c + lhalf*8;
          A[s][c].h[0] = *(const v8h*)(S + kb);
          A[s][c].h[1] = *(const v8h*)(S + kb + 16);
        }
      }

      const _Float16* WL = Wt + layer*HDIM*HDIM;
      const float*    BL = Bs + layer*HDIM;

      for (int nt = 0; nt < 8; ++nt) {
        const int nb = nt * 16;
        const float bv = BL[nb + l16];

        // Preload all 4 k-chunks of the B fragment up front so the in-order
        // DS returns overlap with WMMA issue instead of a hard wait each chunk.
        V16 Bf[4];
        #pragma unroll
        for (int c = 0; c < 4; ++c) {
          const _Float16* Wrow = WL + (nb + l16)*HDIM + 32*c + lhalf*8;
          Bf[c].h[0] = *(const v8h*)(Wrow);
          Bf[c].h[1] = *(const v8h*)(Wrow + 16);
        }

        // All four accumulators start from the inline-0 C operand; bias is
        // added in the element-wise step below (folds into the SiLU chain).
        v8f acc[4];
        #pragma unroll
        for (int r = 0; r < 8; ++r) {
          acc[0][r] = 0.f; acc[1][r] = 0.f; acc[2][r] = 0.f; acc[3][r] = 0.f;
        }
        #pragma unroll
        for (int c = 0; c < 4; ++c) {
          #pragma unroll
          for (int s = 0; s < 4; ++s)
            acc[s] = wmma16(A[s][c].v, Bf[c].v, acc[s]);
        }

        // SiLU chain in D-fragment layout, store f16 to staging for next layer
        #pragma unroll
        for (int r = 0; r < 8; ++r) {
          float z   = acc[0][r] + bv;
          float zx  = acc[1][r];
          float zt  = acc[2][r];
          float zxx = acc[3][r];
          float sg = fsigmoid(z);
          float d1 = sg * (1.f + z*(1.f - sg));
          float d2 = sg * (1.f - sg) * (2.f + z*(1.f - 2.f*sg));
          int m = r + 8*lhalf;
          int col = nb + l16;
          Sh [m*HDIM + col] = (_Float16)(z * sg);
          Sx [m*HDIM + col] = (_Float16)(d1 * zx);
          St [m*HDIM + col] = (_Float16)(d1 * zt);
          Sxx[m*HDIM + col] = (_Float16)(d2*zx*zx + d1*zxx);
        }
      }
    }

    // ---- output layer (128 -> 1): lane pair (l16, l16+16) split K halves ----
    float p0 = 0.f, p1 = 0.f, p2 = 0.f, p3 = 0.f;
    {
      const int m = l16, k0 = lhalf * 64;
      for (int kk = 0; kk < 64; kk += 8) {
        v8h hh = *(const v8h*)(Sh  + m*HDIM + k0 + kk);
        v8h hx = *(const v8h*)(Sx  + m*HDIM + k0 + kk);
        v8h ht = *(const v8h*)(St  + m*HDIM + k0 + kk);
        v8h hq = *(const v8h*)(Sxx + m*HDIM + k0 + kk);
        #pragma unroll
        for (int j = 0; j < 8; ++j) {
          float w = W4s[k0 + kk + j];
          p0 += (float)hh[j] * w;
          p1 += (float)hx[j] * w;
          p2 += (float)ht[j] * w;
          p3 += (float)hq[j] * w;
        }
      }
    }
    float u   = p0 + __shfl_xor(p0, 16, 32) + b4v;
    float ux  = p1 + __shfl_xor(p1, 16, 32);
    float ut  = p2 + __shfl_xor(p2, 16, 32);
    float uxx = p3 + __shfl_xor(p3, 16, 32);

    const int idx = m0 + l16;
    const float xv = PX[wave*16 + l16];
    float r;
    if (MODE == 0) {
      float a  = xv - 0.7f;
      float e  = __builtin_amdgcn_exp2f(-a*a * 50.f * 1.44269504088896340736f); // exp(-(x-0.7)^2/0.02)
      float dS  = -a * 100.f * e;              // S'(x)
      float dS2 = e * (10000.f*a*a - 100.f);   // S''(x)
      r = ut - Dv*uxx + chiv*(ux*dS + u*dS2);
    } else if (MODE == 1) {
      float a = xv - 0.3f;
      r = u - (0.1f + 0.05f * __builtin_amdgcn_exp2f(-a*a * 100.f * 1.44269504088896340736f));
    } else {
      r = ux;
    }
    float contrib = (lhalf == 0 && idx < npts) ? r*r : 0.f;
    #pragma unroll
    for (int off = 16; off > 0; off >>= 1) contrib += __shfl_xor(contrib, off, 32);
    if (lane == 0) atomicAdd(accum, contrib);
  }
}

__global__ void finalize_kernel(const float* __restrict__ ws, float* __restrict__ out) {
  if (threadIdx.x == 0 && blockIdx.x == 0) {
    out[0] = ws[0] / (float)CN_COL
           + ws[1] / (float)CN_IC
           + ws[2] / (float)CN_BC
           + ws[3] / (float)CN_BC;
  }
}

extern "C" void kernel_launch(void* const* d_in, const int* in_sizes, int n_in,
                              void* d_out, int out_size, void* d_ws, size_t ws_size,
                              hipStream_t stream) {
  (void)in_sizes; (void)n_in; (void)out_size; (void)ws_size;
  const float* x_col      = (const float*)d_in[0];
  const float* t_col      = (const float*)d_in[1];
  const float* x_ic       = (const float*)d_in[2];
  const float* t_ic       = (const float*)d_in[3];
  const float* x_bc_left  = (const float*)d_in[4];
  const float* x_bc_right = (const float*)d_in[5];
  const float* t_bc       = (const float*)d_in[6];
  const float* W0 = (const float*)d_in[7];
  const float* b0 = (const float*)d_in[8];
  const float* W1 = (const float*)d_in[9];
  const float* b1 = (const float*)d_in[10];
  const float* W2 = (const float*)d_in[11];
  const float* b2 = (const float*)d_in[12];
  const float* W3 = (const float*)d_in[13];
  const float* b3 = (const float*)d_in[14];
  const float* W4 = (const float*)d_in[15];
  const float* b4 = (const float*)d_in[16];
  const float* Dp = (const float*)d_in[17];
  const float* ch = (const float*)d_in[18];

  float* ws = (float*)d_ws;
  hipMemsetAsync(ws, 0, 4 * sizeof(float), stream);

  (void)hipFuncSetAttribute((const void*)pinn_kernel<0>,
        hipFuncAttributeMaxDynamicSharedMemorySize, (int)SMEM_BYTES);
  (void)hipFuncSetAttribute((const void*)pinn_kernel<1>,
        hipFuncAttributeMaxDynamicSharedMemorySize, (int)SMEM_BYTES);
  (void)hipFuncSetAttribute((const void*)pinn_kernel<2>,
        hipFuncAttributeMaxDynamicSharedMemorySize, (int)SMEM_BYTES);

  pinn_kernel<0><<<256, TPB, SMEM_BYTES, stream>>>(x_col, t_col, CN_COL,
      W0, b0, W1, b1, W2, b2, W3, b3, W4, b4, Dp, ch, ws + 0);
  pinn_kernel<1><<<32, TPB, SMEM_BYTES, stream>>>(x_ic, t_ic, CN_IC,
      W0, b0, W1, b1, W2, b2, W3, b3, W4, b4, Dp, ch, ws + 1);
  pinn_kernel<2><<<32, TPB, SMEM_BYTES, stream>>>(x_bc_left, t_bc, CN_BC,
      W0, b0, W1, b1, W2, b2, W3, b3, W4, b4, Dp, ch, ws + 2);
  pinn_kernel<2><<<32, TPB, SMEM_BYTES, stream>>>(x_bc_right, t_bc, CN_BC,
      W0, b0, W1, b1, W2, b2, W3, b3, W4, b4, Dp, ch, ws + 3);

  finalize_kernel<<<1, 32, 0, stream>>>(ws, (float*)d_out);
}
